// MoELayer_2989297238191
// MI455X (gfx1250) — matrix operand order repaired and checked by reading, not compiled, not run on hardware
//
#include <hip/hip_runtime.h>
#include <math.h>

// Problem constants (from reference): B=4, N=4096, C=1024, F=4096, E=8, K=2
#define T_  16384
#define C_  1024
#define F_  4096
#define E_  8

typedef __attribute__((ext_vector_type(8)))  float          v8f;
typedef __attribute__((ext_vector_type(16))) __bf16         v16bf;
typedef __attribute__((ext_vector_type(16))) unsigned short v16us;
typedef __attribute__((ext_vector_type(8)))  unsigned short v8us;

union Frag16 {
  v16us u;
  v16bf b;
  struct { v8us lo, hi; } h;   // two 16-byte halves -> direct b128 loads
};

__device__ __forceinline__ unsigned short f2bf(float f) {
  unsigned int u = __float_as_uint(f);
  u += 0x7FFFu + ((u >> 16) & 1u);      // round-to-nearest-even
  return (unsigned short)(u >> 16);
}

__device__ __forceinline__ float gelu_exact(float v) {
  return 0.5f * v * (1.0f + erff(v * 0.70710678118654752440f));
}

// ---------------------------------------------------------------------------
// Kernel 0: zero output and per-expert counters
// ---------------------------------------------------------------------------
__global__ void moe_zero(float* __restrict__ out, int* __restrict__ cnt, int n) {
  int gid = blockIdx.x * blockDim.x + threadIdx.x;
  if (gid < 16) cnt[gid] = 0;
  for (int i = gid; i < n; i += gridDim.x * blockDim.x) out[i] = 0.0f;
}

// ---------------------------------------------------------------------------
// Pre-pass A: x fp32 -> bf16 (packed pair stores)
// ---------------------------------------------------------------------------
__global__ __launch_bounds__(256) void cvt_x_bf16(const float* __restrict__ x,
                                                  unsigned short* __restrict__ xbf) {
  const int gid = blockIdx.x * 256 + threadIdx.x;
  const int stride = gridDim.x * 256;
  for (int i = gid; i < T_ * C_ / 2; i += stride) {
    const float a = x[2 * i], b = x[2 * i + 1];
    ((unsigned int*)xbf)[i] = (unsigned int)f2bf(a) | ((unsigned int)f2bf(b) << 16);
  }
}

// ---------------------------------------------------------------------------
// Pre-pass B: per-expert transpose + fp32->bf16  (src [R][Cc] -> dst [Cc][R])
//   makes each lane's WMMA B-fragment K-values two contiguous 8-element runs in memory
// ---------------------------------------------------------------------------
__global__ __launch_bounds__(256) void transpose_cvt_bf16(
    const float* __restrict__ src, unsigned short* __restrict__ dst, int R, int Cc) {
  __shared__ float tile[32][33];
  const int e = blockIdx.z;
  const float* s = src + (size_t)e * R * Cc;
  unsigned short* d = dst + (size_t)e * R * Cc;
  const int bx = blockIdx.x * 32;              // col base in src
  const int by = blockIdx.y * 32;              // row base in src
  const int tx = threadIdx.x, ty = threadIdx.y;  // 32 x 8
#pragma unroll
  for (int k = 0; k < 4; ++k)
    tile[ty + 8 * k][tx] = s[(size_t)(by + ty + 8 * k) * Cc + bx + tx];
  __syncthreads();
#pragma unroll
  for (int k = 0; k < 4; ++k)
    d[(size_t)(bx + ty + 8 * k) * R + by + tx] = f2bf(tile[tx][ty + 8 * k]);
}

// ---------------------------------------------------------------------------
// Kernel 1: router — one wave32 per token; top-2 softmax gates; expert lists
// ---------------------------------------------------------------------------
__global__ __launch_bounds__(256) void moe_router(
    const float* __restrict__ x, const float* __restrict__ rw,
    int* __restrict__ cnt, int* __restrict__ tokList, float* __restrict__ gateList) {
  __shared__ float ldsRW[E_ * C_];                     // 32 KB
  const int tid = threadIdx.x;
  for (int i = tid; i < E_ * C_; i += 256) ldsRW[i] = rw[i];
  __syncthreads();

  const int wave = tid >> 5, lane = tid & 31;
  const int t = blockIdx.x * 8 + wave;                 // T_/8 blocks, exact

  float acc[E_];
#pragma unroll
  for (int e = 0; e < E_; ++e) acc[e] = 0.0f;

  const float* xt = x + (size_t)t * C_;
  for (int c = lane; c < C_; c += 32) {
    const float xv = xt[c];
#pragma unroll
    for (int e = 0; e < E_; ++e) acc[e] += xv * ldsRW[e * C_ + c];
  }
#pragma unroll
  for (int e = 0; e < E_; ++e) {
#pragma unroll
    for (int off = 16; off > 0; off >>= 1) acc[e] += __shfl_xor(acc[e], off);
  }

  if (lane == 0) {
    float mx = acc[0];
#pragma unroll
    for (int e = 1; e < E_; ++e) mx = fmaxf(mx, acc[e]);
    float p[E_], s = 0.0f;
#pragma unroll
    for (int e = 0; e < E_; ++e) { p[e] = __expf(acc[e] - mx); s += p[e]; }
    const float inv = 1.0f / s;
#pragma unroll
    for (int e = 0; e < E_; ++e) p[e] *= inv;

    int i0 = 0; float m0 = p[0];
#pragma unroll
    for (int e = 1; e < E_; ++e) if (p[e] > m0) { m0 = p[e]; i0 = e; }
    int i1 = (i0 == 0) ? 1 : 0; float m1 = p[i1];
#pragma unroll
    for (int e = 0; e < E_; ++e) if (e != i0 && p[e] > m1) { m1 = p[e]; i1 = e; }

    const float rs = 1.0f / (m0 + m1 + 1e-9f);
    const float g0 = m0 * rs, g1 = m1 * rs;

    int p0 = atomicAdd(&cnt[i0], 1);
    tokList[i0 * T_ + p0] = t; gateList[i0 * T_ + p0] = g0;
    int p1 = atomicAdd(&cnt[i1], 1);
    tokList[i1 * T_ + p1] = t; gateList[i1 * T_ + p1] = g1;
  }
}

// ---------------------------------------------------------------------------
// Kernel 2: expert-grouped fused FFN, bf16 WMMA with fp32 accumulation
//   block = 8 waves, 32 tokens (2 M-tiles); wave owns 128 of 1024 out columns
//   B fragments batched into independent registers so loads stay in flight
//   while the matrix pipe drains (partial s_wait_loadcnt instead of 0x0)
// ---------------------------------------------------------------------------
#define LDX 1032   // padded row stride (ushorts) for x tile  (conflict-free ds_b128)
#define LDH 136    // padded row stride (ushorts) for h tile
#define FC  128    // F-chunk width
#define TM  32     // tokens per block

__global__ __launch_bounds__(256) void moe_expert(
    const unsigned short* __restrict__ xbf,   // [T][C]    bf16
    const unsigned short* __restrict__ w1t,   // [E][F][C] bf16 (K-major for GEMM1 B)
    const float* __restrict__ b1,
    const unsigned short* __restrict__ w2t,   // [E][C][F] bf16 (K-major for GEMM2 B)
    const float* __restrict__ b2,
    const int* __restrict__ cnt, const int* __restrict__ tokList,
    const float* __restrict__ gateList, float* __restrict__ out) {
  const int e    = blockIdx.y;
  const int base = blockIdx.x * TM;
  const int nTok = cnt[e];
  if (base >= nTok) return;

  __shared__ __align__(16) unsigned short ldsX[TM * LDX];  // 66 KB bf16 x tile
  __shared__ __align__(16) unsigned short ldsH[TM * LDH];  // 8.7 KB bf16 h tile
  __shared__ int   ldsTok[TM];
  __shared__ float ldsGate[TM];

  const int tid = threadIdx.x;
  if (tid < TM) {
    const int j = base + tid;
    if (j < nTok) { ldsTok[tid] = tokList[e * T_ + j]; ldsGate[tid] = gateList[e * T_ + j]; }
    else          { ldsTok[tid] = tokList[e * T_ + base]; ldsGate[tid] = 0.0f; } // pad: gate 0
  }
  __syncthreads();

  // Stage x tile [TM][C] bf16 -> LDS as pure 16-byte copies
  for (int i = tid; i < TM * (C_ / 8); i += 256) {
    const int m = i >> 7, c8 = (i & 127) * 8;
    *(v8us*)&ldsX[m * LDX + c8] = *(const v8us*)&xbf[(size_t)ldsTok[m] * C_ + c8];
  }
  __syncthreads();

  const int wv = tid >> 5, lane = tid & 31;
  const int lm = lane & 15, half = lane >> 4;
  const unsigned short* w1e = w1t + (size_t)e * F_ * C_;
  const unsigned short* w2e = w2t + (size_t)e * F_ * C_;

  v8f acc[2][8];
#pragma unroll
  for (int mt = 0; mt < 2; ++mt)
#pragma unroll
    for (int nt = 0; nt < 8; ++nt) acc[mt][nt] = (v8f){0,0,0,0,0,0,0,0};

  for (int f0 = 0; f0 < F_; f0 += FC) {
    // ---- GEMM1: h[TM][FC] = x[TM][C] @ w1[:, f0:f0+FC]; wave wv -> cols 16*wv..+15
    v8f hacc0 = (v8f){0,0,0,0,0,0,0,0};
    v8f hacc1 = (v8f){0,0,0,0,0,0,0,0};
    const int col1 = f0 + 16 * wv + lm;
    const unsigned short* w1row = w1e + (size_t)col1 * C_;   // contiguous over K
    for (int k0 = 0; k0 < C_; k0 += 64) {                    // 2 k-steps per iter
      Frag16 a0[2], a1[2], bb[2];
#pragma unroll
      for (int kk = 0; kk < 2; ++kk) {
        const int ko = k0 + 32 * kk + 8 * half;
        const int kb = k0 + 32 * kk + 8 * half;
        bb[kk].h.lo = *(const v8us*)&w1row[kb];
        bb[kk].h.hi = *(const v8us*)&w1row[kb + 16];
        a0[kk].h.lo = *(const v8us*)&ldsX[lm * LDX + ko];
        a0[kk].h.hi = *(const v8us*)&ldsX[lm * LDX + ko + 16];
        a1[kk].h.lo = *(const v8us*)&ldsX[(16 + lm) * LDX + ko];
        a1[kk].h.hi = *(const v8us*)&ldsX[(16 + lm) * LDX + ko + 16];
      }
#pragma unroll
      for (int kk = 0; kk < 2; ++kk) {
        hacc0 = __builtin_amdgcn_wmma_f32_16x16x32_bf16(false, a0[kk].b, false, bb[kk].b,
                                                        (short)0, hacc0, false, false);
        hacc1 = __builtin_amdgcn_wmma_f32_16x16x32_bf16(false, a1[kk].b, false, bb[kk].b,
                                                        (short)0, hacc1, false, false);
      }
    }
    // bias + exact GELU -> ldsH (bf16, A-operand for GEMM2)
    const float b1v = b1[(size_t)e * F_ + col1];
    const int   hcol = 16 * wv + lm;
#pragma unroll
    for (int r = 0; r < 8; ++r) {
      const int m = r + 8 * half;
      ldsH[m * LDH + hcol]        = f2bf(gelu_exact(hacc0[r] + b1v));
      ldsH[(16 + m) * LDH + hcol] = f2bf(gelu_exact(hacc1[r] + b1v));
    }
    __syncthreads();

    // ---- GEMM2 partial: y[TM][1024] += h[TM][FC] @ w2[f0:f0+FC, :]
    for (int ks = 0; ks < FC; ks += 32) {
      Frag16 a0, a1;
      const int ko = ks + 8 * half;
      a0.h.lo = *(const v8us*)&ldsH[lm * LDH + ko];
      a0.h.hi = *(const v8us*)&ldsH[lm * LDH + ko + 16];
      a1.h.lo = *(const v8us*)&ldsH[(16 + lm) * LDH + ko];
      a1.h.hi = *(const v8us*)&ldsH[(16 + lm) * LDH + ko + 16];
      const int kb = f0 + ks + 8 * half;
      const unsigned short* w2base = w2e + (size_t)(128 * wv + lm) * F_ + kb;
      // load all 8 B fragments into independent registers (one deep clause)
      Frag16 bb[8];
#pragma unroll
      for (int nt = 0; nt < 8; ++nt) {
        const unsigned short* w2row = w2base + (size_t)(16 * nt) * F_;
        bb[nt].h.lo = *(const v8us*)&w2row[0];
        bb[nt].h.hi = *(const v8us*)&w2row[16];
      }
      // drain the matrix pipe while later loads are still in flight
#pragma unroll
      for (int nt = 0; nt < 8; ++nt) {
        acc[0][nt] = __builtin_amdgcn_wmma_f32_16x16x32_bf16(false, a0.b, false, bb[nt].b,
                                                             (short)0, acc[0][nt], false, false);
        acc[1][nt] = __builtin_amdgcn_wmma_f32_16x16x32_bf16(false, a1.b, false, bb[nt].b,
                                                             (short)0, acc[1][nt], false, false);
      }
    }
    __syncthreads();   // ldsH reused next chunk
  }

  // ---- epilogue: +b2, *gate, scatter-add into out (token may have 2 experts)
  const float* b2e = b2 + (size_t)e * C_;
#pragma unroll
  for (int nt = 0; nt < 8; ++nt) {
    const int col = 128 * wv + 16 * nt + lm;
    const float b2v = b2e[col];
#pragma unroll
    for (int mt = 0; mt < 2; ++mt) {
#pragma unroll
      for (int r = 0; r < 8; ++r) {
        const int m = 16 * mt + r + 8 * half;
        const float y = acc[mt][nt][r] + b2v;
        atomicAdd(&out[(size_t)ldsTok[m] * C_ + col], ldsGate[m] * y);
      }
    }
  }
}

// ---------------------------------------------------------------------------
extern "C" void kernel_launch(void* const* d_in, const int* in_sizes, int n_in,
                              void* d_out, int out_size, void* d_ws, size_t ws_size,
                              hipStream_t stream) {
  const float* x  = (const float*)d_in[0];
  const float* rw = (const float*)d_in[1];
  const float* w1 = (const float*)d_in[2];
  const float* b1 = (const float*)d_in[3];
  const float* w2 = (const float*)d_in[4];
  const float* b2 = (const float*)d_in[5];
  float* out = (float*)d_out;

  // workspace layout (~169 MB):
  //   [256B cnt][E*T int tokList][E*T f32 gateList][xbf 33.5MB][w1t 67MB][w2t 67MB]
  char* ws = (char*)d_ws;
  int*   cnt      = (int*)ws;
  int*   tokList  = (int*)(ws + 256);
  float* gateList = (float*)(ws + 256 + (size_t)E_ * T_ * 4);
  unsigned short* xbf = (unsigned short*)(ws + 256 + (size_t)E_ * T_ * 8);
  unsigned short* w1t = xbf + (size_t)T_ * C_;
  unsigned short* w2t = w1t + (size_t)E_ * F_ * C_;

  moe_zero<<<1024, 256, 0, stream>>>(out, cnt, T_ * C_);
  cvt_x_bf16<<<2048, 256, 0, stream>>>(x, xbf);
  dim3 tb(32, 8);
  transpose_cvt_bf16<<<dim3(F_ / 32, C_ / 32, E_), tb, 0, stream>>>(w1, w1t, C_, F_);
  transpose_cvt_bf16<<<dim3(C_ / 32, F_ / 32, E_), tb, 0, stream>>>(w2, w2t, F_, C_);
  moe_router<<<T_ / 8, 256, 0, stream>>>(x, rw, cnt, tokList, gateList);
  dim3 grid(T_ / TM, E_);
  moe_expert<<<grid, 256, 0, stream>>>(xbf, w1t, b1, w2t, b2, cnt, tokList, gateList, out);
}
